// ConvLayer_472446403130
// MI455X (gfx1250) — matrix operand order repaired and checked by reading, not compiled, not run on hardware
//
#include <hip/hip_runtime.h>
#include <cstdint>

// Problem constants (from reference): B=8, N=1024, T=8, Cin=Cout=64, sigma=1
#define BB   8
#define NN   1024
#define TT   8
#define CIN  64
#define COUT 64

typedef __bf16 v16bf __attribute__((ext_vector_type(16)));
typedef float  v8f   __attribute__((ext_vector_type(8)));

__device__ __forceinline__ uint16_t f2bf(float x) {
    uint32_t b = __float_as_uint(x);
    // round-to-nearest-even truncation to bf16
    return (uint16_t)((b + 0x7FFFu + ((b >> 16) & 1u)) >> 16);
}

// ---------------------------------------------------------------------------
// Phase 1: per point n of batch b:
//   rowsum[b,n]   = sum_m exp(-dist(n,m)/2)
//   a[(b*T+t),n]  = exp(p_n·t - |t|^2/2)     (post-GEMM row scale)
//   c[(b*T+t),n]  = exp(-p_n·t)              (folded into H)
// ---------------------------------------------------------------------------
__global__ void __launch_bounds__(256)
phase1_point(const float* __restrict__ points, const float* __restrict__ trans,
             float* __restrict__ rowsum, float* __restrict__ aArr,
             float* __restrict__ cArr) {
    __shared__ float pS[NN * 3];
    __shared__ float sqS[NN];
    const int b = blockIdx.y;
    const int tid = threadIdx.x;
    const float* pb = points + (size_t)b * NN * 3;
    for (int i = tid; i < NN * 3; i += 256) pS[i] = pb[i];
    __syncthreads();
    for (int i = tid; i < NN; i += 256) {
        float x = pS[3 * i], y = pS[3 * i + 1], z = pS[3 * i + 2];
        sqS[i] = x * x + y * y + z * z;
    }
    __syncthreads();

    const int n = blockIdx.x * 256 + tid;
    const float px = pS[3 * n], py = pS[3 * n + 1], pz = pS[3 * n + 2];
    const float sqn = sqS[n];
    float s = 0.f;
    for (int m = 0; m < NN; ++m) {
        float d = sqn + sqS[m] -
                  2.f * (px * pS[3 * m] + py * pS[3 * m + 1] + pz * pS[3 * m + 2]);
        s += __expf(-0.5f * d);
    }
    rowsum[b * NN + n] = s;

#pragma unroll
    for (int t = 0; t < TT; ++t) {
        float tx = trans[(b * TT + t) * 3 + 0];
        float ty = trans[(b * TT + t) * 3 + 1];
        float tz = trans[(b * TT + t) * 3 + 2];
        float ptd = px * tx + py * ty + pz * tz;
        float tsq = tx * tx + ty * ty + tz * tz;
        aArr[(size_t)(b * TT + t) * NN + n] = __expf(ptd - 0.5f * tsq);
        cArr[(size_t)(b * TT + t) * NN + n] = __expf(-ptd);
    }
}

// ---------------------------------------------------------------------------
// Phase 2: H[((b*T+t)*64 + o)*N + m] = c[t,b,m] * sum_cin w[b,m,cin]*k[o,cin,t] / rowsum[b,m]
// stored bf16, m-contiguous (so 32x16 B tiles load row-pairs as dwords).
// One block per (b,t,o); k column cached in LDS.
// ---------------------------------------------------------------------------
__global__ void __launch_bounds__(256)
phase2_h(const float* __restrict__ functions, const float* __restrict__ k_tensor,
         const float* __restrict__ rowsum, const float* __restrict__ cArr,
         uint16_t* __restrict__ H) {
    const int bo = blockIdx.x;      // (b*T+t)*64 + o
    const int o  = bo & 63;
    const int bt = bo >> 6;         // b*T + t
    const int b  = bt >> 3;
    const int t  = bt & 7;
    __shared__ float kS[CIN];
    if (threadIdx.x < CIN)
        kS[threadIdx.x] = k_tensor[((size_t)o * CIN + threadIdx.x) * TT + t];
    __syncthreads();

#pragma unroll
    for (int mi = 0; mi < 4; ++mi) {
        int m = threadIdx.x + mi * 256;
        const float* f = functions + ((size_t)b * NN + m) * CIN;
        float acc = 0.f;
#pragma unroll
        for (int c = 0; c < CIN; ++c) acc += f[c] * kS[c];
        float val = acc / rowsum[b * NN + m] * cArr[(size_t)bt * NN + m];
        H[(size_t)bo * NN + m] = f2bf(val);
    }
}

// ---------------------------------------------------------------------------
// Phase 3: main GEMM.  Block = one 16-row n-tile of batch b, 8 waves.
// Wave w handles translation t=w over all 64 output columns (4 col-tiles).
// K tile (16x32 bf16) is built cooperatively in LDS and DOUBLE-BUFFERED:
// each step reads buf[i&1] while building buf[(i+1)&1], so only ONE barrier
// per K-step, and the exp/VALU tile construction co-executes with WMMA.
// acc_t is scaled by a[t,n] and reduced across waves through LDS.
// ---------------------------------------------------------------------------
__global__ void __launch_bounds__(256)
phase3_gemm(const float* __restrict__ points, const uint16_t* __restrict__ H,
            const float* __restrict__ aArr, float* __restrict__ out_pre) {
    __shared__ float    pS[NN * 3];
    __shared__ float    sqS[NN];
    __shared__ uint16_t ktS[2][16 * 32];  // double-buffered 16(n) x 32(m) bf16 tile
    __shared__ float    aS[TT * 16];
    __shared__ float    redS[8 * 16 * 64];

    const int b    = blockIdx.y;
    const int n0   = blockIdx.x * 16;
    const int tid  = threadIdx.x;
    const int lane = tid & 31;
    const int wave = tid >> 5;            // == translation t

    const float* pb = points + (size_t)b * NN * 3;
    for (int i = tid; i < NN * 3; i += 256) pS[i] = pb[i];
    __syncthreads();
    for (int i = tid; i < NN; i += 256) {
        float x = pS[3 * i], y = pS[3 * i + 1], z = pS[3 * i + 2];
        sqS[i] = x * x + y * y + z * z;
    }
    if (tid < TT * 16)
        aS[tid] = aArr[((size_t)b * TT + (tid >> 4)) * NN + n0 + (tid & 15)];
    __syncthreads();

    // Each thread owns K-tile entries (row=wave, col=lane) and (row=wave+8, col=lane)
    const int nA = n0 + wave, nB = n0 + wave + 8;
    const float ax = pS[3 * nA], ay = pS[3 * nA + 1], az = pS[3 * nA + 2], asq = sqS[nA];
    const float bx = pS[3 * nB], by = pS[3 * nB + 1], bz = pS[3 * nB + 2], bsq = sqS[nB];

    v8f acc[4] = {v8f{}, v8f{}, v8f{}, v8f{}};
    const uint16_t* Hbt = H + (size_t)(b * TT + wave) * COUT * NN;

    const int arow = lane & 15;
    const int kofs = (lane >= 16) ? 8 : 0;
    const int rsel = (lane >= 16) ? 8 : 0;
    const int ocol = lane & 15;

    // Prologue: build tile 0 into buffer 0.
    {
        int m = lane;
        float qx = pS[3 * m], qy = pS[3 * m + 1], qz = pS[3 * m + 2];
        float sqm = sqS[m];
        float d0 = asq + sqm - 2.f * (ax * qx + ay * qy + az * qz);
        float d1 = bsq + sqm - 2.f * (bx * qx + by * qy + bz * qz);
        ktS[0][wave * 32 + lane]       = f2bf(__expf(-0.5f * d0));
        ktS[0][(wave + 8) * 32 + lane] = f2bf(__expf(-0.5f * d1));
    }

    for (int it = 0; it < NN / 32; ++it) {
        __syncthreads();                  // publish buf[it&1]; retire buf[(it+1)&1] readers
        const int m0  = it * 32;
        const int cur = it & 1;
        const int nxt = cur ^ 1;

        // A fragment from current buffer: 16-bit A 16x32 layout (ISA 7.12.2):
        // lane<16 -> M=lane, VGPR j: K = 2j,2j+1 (+8 for lanes>=16, +16 for j>=4)
        union { uint32_t u[8]; v16bf v; } A;
        const uint32_t* kt32 = (const uint32_t*)ktS[cur];   // 16 dwords per row
#pragma unroll
        for (int j = 0; j < 8; ++j) {
            int kk = ((j < 4) ? (2 * j) : (16 + 2 * (j - 4))) + kofs;
            A.u[j] = kt32[arow * 16 + (kk >> 1)];
        }

        // Build NEXT tile into the other buffer (wrapped on the last step —
        // branchless, keeps EXEC all-ones; redundant final write is harmless).
        {
            int m = ((m0 + 32) & (NN - 1)) + lane;
            float qx = pS[3 * m], qy = pS[3 * m + 1], qz = pS[3 * m + 2];
            float sqm = sqS[m];
            float d0 = asq + sqm - 2.f * (ax * qx + ay * qy + az * qz);
            float d1 = bsq + sqm - 2.f * (bx * qx + by * qy + bz * qz);
            ktS[nxt][wave * 32 + lane]       = f2bf(__expf(-0.5f * d0));
            ktS[nxt][(wave + 8) * 32 + lane] = f2bf(__expf(-0.5f * d1));
        }

        // B fragments (32x16): lane col = lane%16, VGPR j rows = K = 2j,2j+1 (+8 for lanes>=16, +16 for j>=4)
#pragma unroll
        for (int c = 0; c < 4; ++c) {
            union { uint32_t u[8]; v16bf v; } Bf;
            const uint16_t* col = Hbt + (size_t)(c * 16 + ocol) * NN + m0 + rsel;
#pragma unroll
            for (int j = 0; j < 8; ++j)
                Bf.u[j] = *(const uint32_t*)(col + ((j < 4) ? (2 * j) : (16 + 2 * (j - 4))));
            acc[c] = __builtin_amdgcn_wmma_f32_16x16x32_bf16(
                false, A.v, false, Bf.v, (short)0, acc[c], false, false);
        }
    }

    // Scale by a[t, n] and stage per-wave partials in LDS.
    {
        const int rowadd = (lane >= 16) ? 8 : 0;
#pragma unroll
        for (int c = 0; c < 4; ++c)
#pragma unroll
            for (int r = 0; r < 8; ++r) {
                int row = r + rowadd;     // C/D layout: M = r + 8*(lane>=16)
                redS[wave * 1024 + row * 64 + c * 16 + ocol] =
                    acc[c][r] * aS[wave * 16 + row];
            }
    }
    __syncthreads();

    // Reduce over the 8 waves (= sum over translations t)
#pragma unroll
    for (int k = 0; k < 4; ++k) {
        int idx = tid * 4 + k;            // 0..1023 covers 16x64 tile
        float s = 0.f;
#pragma unroll
        for (int w = 0; w < 8; ++w) s += redS[w * 1024 + idx];
        int row = idx >> 6, o = idx & 63;
        out_pre[((size_t)b * NN + n0 + row) * COUT + o] = s;
    }
}

// ---------------------------------------------------------------------------
// Phase 4: BatchNorm (training stats over B*N) + ReLU.  One block per channel.
// ---------------------------------------------------------------------------
__global__ void __launch_bounds__(256)
phase4_bn(const float* __restrict__ out_pre, const float* __restrict__ gamma,
          const float* __restrict__ beta, float* __restrict__ out) {
    const int o = blockIdx.x;
    const int tid = threadIdx.x;
    __shared__ float sS[256], s2S[256];
    float s = 0.f, s2 = 0.f;
    for (int i = tid; i < BB * NN; i += 256) {
        float v = out_pre[(size_t)i * COUT + o];
        s += v; s2 += v * v;
    }
    sS[tid] = s; s2S[tid] = s2;
    __syncthreads();
    for (int st = 128; st > 0; st >>= 1) {
        if (tid < st) { sS[tid] += sS[tid + st]; s2S[tid] += s2S[tid + st]; }
        __syncthreads();
    }
    const float inv  = 1.0f / (float)(BB * NN);
    const float mean = sS[0] * inv;
    const float var  = s2S[0] * inv - mean * mean;
    const float scl  = rsqrtf(var + 1e-5f) * gamma[o];
    const float sh   = beta[o];
    for (int i = tid; i < BB * NN; i += 256) {
        float v = (out_pre[(size_t)i * COUT + o] - mean) * scl + sh;
        out[(size_t)i * COUT + o] = fmaxf(v, 0.f);
    }
}

// ---------------------------------------------------------------------------
extern "C" void kernel_launch(void* const* d_in, const int* in_sizes, int n_in,
                              void* d_out, int out_size, void* d_ws, size_t ws_size,
                              hipStream_t stream) {
    const float* points    = (const float*)d_in[0];   // [8,1024,3]
    const float* trans     = (const float*)d_in[1];   // [8,8,3]
    const float* functions = (const float*)d_in[2];   // [8,1024,64]
    const float* k_tensor  = (const float*)d_in[3];   // [64,64,8]
    const float* gamma     = (const float*)d_in[4];   // [64]
    const float* beta      = (const float*)d_in[5];   // [64]
    float* out = (float*)d_out;                       // [8,1024,64] fp32

    char* ws = (char*)d_ws;
    float*    rowsum  = (float*)(ws + 0);                         //  32 KB
    float*    aArr    = (float*)(ws + 32768);                     // 256 KB
    float*    cArr    = (float*)(ws + 32768 + 262144);            // 256 KB
    uint16_t* H       = (uint16_t*)(ws + 557056);                 //   8 MB bf16
    float*    out_pre = (float*)(ws + 8945664);                   //   2 MB

    phase1_point<<<dim3(NN / 256, BB), 256, 0, stream>>>(points, trans, rowsum, aArr, cArr);
    phase2_h<<<dim3(BB * TT * COUT), 256, 0, stream>>>(functions, k_tensor, rowsum, cArr, H);
    phase3_gemm<<<dim3(NN / 16, BB), 256, 0, stream>>>(points, H, aArr, out_pre);
    phase4_bn<<<dim3(COUT), 256, 0, stream>>>(out_pre, gamma, beta, out);
}
